// Model_37993280700890
// MI455X (gfx1250) — matrix operand order repaired and checked by reading, not compiled, not run on hardware
//
#include <hip/hip_runtime.h>

typedef _Float16 f16;
typedef __attribute__((ext_vector_type(16))) _Float16 v16h;
typedef __attribute__((ext_vector_type(16))) float    v16f;
typedef __attribute__((ext_vector_type(8)))  float    v8f;

#define DD 43
#define ROWS_PER_TILE 128               // 8 waves x 16 rows
#define NTILES_TOTAL  2048              // 262144 / 128
#define TILES_PER_BLOCK 4
#define NBLOCKS (NTILES_TOTAL / TILES_PER_BLOCK)

// Combined (padded) K layout, K = 96:
//   k in [0,43)   -> x[b][k]
//   k in [43,48)  -> 0
//   k in [48,91)  -> h[b][k-48]
//   k == 91       -> 1.0   (bias column; weight row holds bx+bh)
//   k in [92,96)  -> 0
#define KH   48                          // h segment base
#define KB   (KH + DD)                   // bias column = 91

#define WSLOTS (4*3*3*32)                // v16h slots: gates * ntiles * kchunks * lanes
#define ASLOTS (8*3*32)                  // slots: waves * kchunks * lanes

// ---- CDNA5 async global->LDS path (guarded; falls back to cvt staging) ----
#if __has_builtin(__builtin_amdgcn_global_load_async_to_lds_b32) && \
    __has_builtin(__builtin_amdgcn_s_wait_asynccnt)
#define USE_ASYNC 1
typedef __attribute__((address_space(1))) int gi32;   // global int*
typedef __attribute__((address_space(3))) int li32;   // LDS int*
#else
#define USE_ASYNC 0
#endif

// ---- transcendentals: prefer the CDNA5 trans-pipe tanh if the builtin exists
__device__ __forceinline__ float tanh_exp(float v) {
    float t = __expf(-2.0f * v);
    return (1.0f - t) / (1.0f + t);
}
#if __has_builtin(__builtin_amdgcn_tanhf)
#define TANHF(v) __builtin_amdgcn_tanhf(v)
#elif __has_builtin(__builtin_amdgcn_tanh_f32)
#define TANHF(v) __builtin_amdgcn_tanh_f32(v)
#else
#define TANHF(v) tanh_exp(v)
#endif
__device__ __forceinline__ float fast_sigmoid(float v) {
    return __builtin_fmaf(0.5f, TANHF(0.5f * v), 0.5f);
}

__global__ __launch_bounds__(256) void lstm_fused(
    const float* __restrict__ x,  const float* __restrict__ h,
    const float* __restrict__ c,
    const float* __restrict__ Wx, const float* __restrict__ bx,
    const float* __restrict__ Wh, const float* __restrict__ bh,
    float* __restrict__ out)
{
    __shared__ __align__(32) f16 w_lds[WSLOTS * 16];   // B operand, WMMA reg layout
#if USE_ASYNC
    __shared__ __align__(64) float a_st[ASLOTS * 16];  // A operand, f32, async-filled
#else
    __shared__ __align__(32) f16   a_st[ASLOTS * 16];  // A operand, f16, cvt-staged
#endif

    const int tid  = threadIdx.x;
    const int lane = tid & 31;
    const int wave = tid >> 5;

    // ================= weight staging: ONCE per block (amortized x4) ========
    // element (g, nt, ch, lane, e):  k = ch*32 + (lane>=16)*8 + (e<8 ? e : e+8)
    //                                o = nt*16 + (lane&15)
    {
        const int o_lo = lane & 15;
        const int ehi  = (tid >> 5) << 1;          // 0,2,..,14
        #pragma unroll
        for (int g = 0; g < 4; ++g)
        #pragma unroll
        for (int nt = 0; nt < 3; ++nt)
        #pragma unroll
        for (int ch = 0; ch < 3; ++ch)
        #pragma unroll
        for (int j2 = 0; j2 < 2; ++j2) {
            int e = ehi + j2;
            int k = (ch << 5) + ((lane >> 4) << 3) + (e < 8 ? e : e + 8);
            int o = (nt << 4) + o_lo;
            float v = 0.0f;
            if (o < DD) {
                if (k < DD)                 v = Wx[(g*DD + o)*DD + k];
                else if (k >= KH && k < KB) v = Wh[(g*DD + o)*DD + (k - KH)];
                else if (k == KB)           v = bx[g*DD + o] + bh[g*DD + o];
            }
            w_lds[((((g*3 + nt)*3 + ch) << 5) + lane) * 16 + e] = (f16)v;
        }
    }

    // A staging: thread supplies (e, row_lo); loop supplies (wave-tile, chunk, lane_hi)
    const int e_a  = tid & 15;
    const int rl   = (tid >> 4) & 15;
    const int eoff = (e_a < 8) ? e_a : (e_a + 8);   // A layout: elems 8..15 -> K+16..23

#if USE_ASYNC
    // Pad/bias columns are constants: write once; async loads never touch them.
    #pragma unroll
    for (int tw = 0; tw < 8; ++tw) {
        int k1 = 40 + eoff;                 // (ch=1, lh=1) run
        if (k1 >= DD && k1 < KH)
            a_st[((((tw*3 + 1) << 5) + 16 + rl)) * 16 + e_a] = 0.0f;
        int k2 = 72 + eoff;                 // (ch=2, lh=1) run
        if (k2 >= KB)
            a_st[((((tw*3 + 2) << 5) + 16 + rl)) * 16 + e_a] =
                (k2 == KB) ? 1.0f : 0.0f;
    }
#endif

    const v16h* wop = (const v16h*)w_lds;
#if USE_ASYNC
    const v16f* aop = (const v16f*)a_st;
#else
    const v16h* aop = (const v16h*)a_st;
#endif
    const int n0  = lane & 15;
    const int mhi = (lane >> 4) << 3;

    #pragma unroll 1
    for (int it = 0; it < TILES_PER_BLOCK; ++it) {
        const int tile  = blockIdx.x * TILES_PER_BLOCK + it;
        const int tRow0 = tile * ROWS_PER_TILE;

        __syncthreads();   // previous tile fully consumed (and weights/consts ready)

        // ---- stage activations in A-operand register layout ----------------
        #pragma unroll
        for (int tw = 0; tw < 8; ++tw)
        #pragma unroll
        for (int ch = 0; ch < 3; ++ch)
        #pragma unroll
        for (int lh = 0; lh < 2; ++lh) {
            int k = (ch << 5) + (lh << 3) + eoff;   // lanes>=16 hold K+8 / K+24
            int b = tRow0 + (tw << 4) + rl;
#if USE_ASYNC
            float* lp = &a_st[((((tw*3 + ch) << 5) + (lh << 4) + rl)) * 16 + e_a];
            if (k < DD)
                __builtin_amdgcn_global_load_async_to_lds_b32(
                    (gi32*)const_cast<float*>(x + b*DD + k), (li32*)lp, 0, 0);
            else if (k >= KH && k < KB)
                __builtin_amdgcn_global_load_async_to_lds_b32(
                    (gi32*)const_cast<float*>(h + b*DD + (k - KH)), (li32*)lp, 0, 0);
            // else: constant slot, pre-written once per block
#else
            float v;
            if (k < DD)            v = x[b*DD + k];
            else if (k < KH)       v = 0.0f;
            else if (k < KB)       v = h[b*DD + (k - KH)];
            else if (k == KB)      v = 1.0f;
            else                   v = 0.0f;
            a_st[((((tw*3 + ch) << 5) + (lh << 4) + rl)) * 16 + e_a] = (f16)v;
#endif
        }

#if USE_ASYNC
        __builtin_amdgcn_s_wait_asynccnt(0);   // my async copies landed in LDS
#endif
        __syncthreads();                        // everyone's copies landed

        // ---- gather A operand: contiguous bytes per lane -------------------
        v16h a[3];
        #pragma unroll
        for (int ch = 0; ch < 3; ++ch) {
#if USE_ASYNC
            v16f af = aop[(wave*3 + ch)*32 + lane];
            a[ch] = __builtin_convertvector(af, v16h);
#else
            a[ch] = aop[(wave*3 + ch)*32 + lane];
#endif
        }

        // ---- prefetch c (clamped column for padded lanes) ------------------
        const int brow = tRow0 + (wave << 4) + mhi;
        float cv[3][8];
        #pragma unroll
        for (int nt = 0; nt < 3; ++nt) {
            int o  = nt*16 + n0;
            int oc = (o < DD) ? o : (DD - 1);
            #pragma unroll
            for (int r = 0; r < 8; ++r)
                cv[nt][r] = c[(brow + r)*DD + oc];
        }

        // ---- fused GEMM: 4 gates x 3 N-tiles x 3 K-chunks = 36 WMMAs -------
        v8f acc[4][3] = {};
        #pragma unroll
        for (int g = 0; g < 4; ++g)
        #pragma unroll
        for (int nt = 0; nt < 3; ++nt)
        #pragma unroll
        for (int ch = 0; ch < 3; ++ch) {
            v16h bm = wop[((g*3 + nt)*3 + ch)*32 + lane];
            acc[g][nt] = __builtin_amdgcn_wmma_f32_16x16x32_f16(
                false, a[ch], false, bm, (short)0, acc[g][nt], false, false);
        }

        // ---- fused elementwise LSTM (C/D layout: VGPR r -> M=r+8*(lane>=16),
        //      N = lane&15) --------------------------------------------------
        #pragma unroll
        for (int nt = 0; nt < 3; ++nt) {
            int o = nt*16 + n0;
            if (o < DD) {
                #pragma unroll
                for (int r = 0; r < 8; ++r) {
                    float ig = fast_sigmoid(acc[0][nt][r]);
                    float fg = fast_sigmoid(acc[1][nt][r]);
                    float gg = TANHF(acc[2][nt][r]);
                    float og = fast_sigmoid(acc[3][nt][r]);
                    float cn = __builtin_fmaf(fg, cv[nt][r], ig * gg);
                    out[(brow + r)*DD + o] = og * TANHF(cn);
                }
            }
        }
    }
}

extern "C" void kernel_launch(void* const* d_in, const int* in_sizes, int n_in,
                              void* d_out, int out_size, void* d_ws, size_t ws_size,
                              hipStream_t stream) {
    const float* x  = (const float*)d_in[0];
    const float* h  = (const float*)d_in[1];
    const float* c  = (const float*)d_in[2];
    const float* Wx = (const float*)d_in[3];
    const float* bx = (const float*)d_in[4];
    const float* Wh = (const float*)d_in[5];
    const float* bh = (const float*)d_in[6];
    float* out = (float*)d_out;

    dim3 grid(NBLOCKS);
    dim3 block(256);
    hipLaunchKernelGGL(lstm_fused, grid, block, 0, stream,
                       x, h, c, Wx, bx, Wh, bh, out);
}
